// Transformer_17179869207
// MI455X (gfx1250) — matrix operand order repaired and checked by reading, not compiled, not run on hardware
//
#include <hip/hip_runtime.h>
#include <hip/hip_bf16.h>

// ---------------- problem constants ----------------
#define TDIM 1024
#define TDEPTH 6
#define THEADS 16
#define THEAD_DIM 64
#define THID 2048
#define TB 4
#define TSEQ 1024
#define TROWS (TB * TSEQ)   // 4096

typedef __attribute__((ext_vector_type(16))) __bf16 v16bf;
typedef __attribute__((ext_vector_type(8)))  __bf16 v8bf;
typedef __attribute__((ext_vector_type(8)))  float  v8f;
typedef unsigned u32x4 __attribute__((ext_vector_type(4)));
typedef int      i32x8 __attribute__((ext_vector_type(8)));
typedef int      i32x4 __attribute__((ext_vector_type(4)));

// fp32 -> bf16 RNE (integer path; cold/epilogue code only)
__device__ __forceinline__ __bf16 f2bf(float f) {
    union { float f; unsigned u; } a; a.f = f;
    unsigned r = a.u + 0x7FFFu + ((a.u >> 16) & 1u);
    unsigned short s = (unsigned short)(r >> 16);
    __bf16 h; __builtin_memcpy(&h, &s, 2);
    return h;
}

// two aligned 16B loads (LDS or global) -> one 16-element bf16 fragment
__device__ __forceinline__ v16bf ld_frag(const __bf16* p0, const __bf16* p1) {
    v8bf a = *(const v8bf*)p0;
    v8bf b = *(const v8bf*)p1;
    return __builtin_shufflevector(a, b, 0,1,2,3,4,5,6,7,8,9,10,11,12,13,14,15);
}

__device__ __forceinline__ v8f zero8() {
    v8f z;
#pragma unroll
    for (int i = 0; i < 8; i++) z[i] = 0.0f;
    return z;
}

#define WMMA_BF16(A, B, C) \
    __builtin_amdgcn_wmma_f32_16x16x32_bf16(false, (A), false, (B), (short)0, (C), false, false)

// ==========================================================================
// Tensor Data Mover: 2-D tile (rows of bf16) global -> LDS with row padding.
// ==========================================================================
#if __has_builtin(__builtin_amdgcn_tensor_load_to_lds) && \
    __has_builtin(__builtin_amdgcn_s_wait_tensorcnt)
#define HAS_TDM 1
#else
#define HAS_TDM 0
#endif

#if HAS_TDM
__device__ __forceinline__ void tdm_load_2d(unsigned lds_off, const __bf16* gsrc,
                                            unsigned tensor_d0, unsigned tensor_d1,
                                            unsigned tile_d0, unsigned tile_d1,
                                            unsigned pad_int_code, unsigned pad_amt_code)
{
    unsigned long long ga = (unsigned long long)(size_t)gsrc;
    u32x4 g0;
    g0[0] = 1u;                                    // count=1, user descriptor
    g0[1] = lds_off;                               // lds_addr (bytes)
    g0[2] = (unsigned)(ga & 0xffffffffu);          // global_addr[31:0]
    g0[3] = (unsigned)((ga >> 32) & 0x01ffffffu)   // global_addr[56:32]
          | (2u << 30);                            // type = 2 ("image")
    i32x8 g1;
    g1[0] = (int)((1u << 16)                       // data_size = 2 bytes
                | (1u << 20)                       // pad_enable
                | (pad_int_code << 22)
                | (pad_amt_code << 25));
    g1[1] = (int)((tensor_d0 & 0xffffu) << 16);                          // dim0 lo16
    g1[2] = (int)((tensor_d0 >> 16) | ((tensor_d1 & 0xffffu) << 16));    // dim0 hi | dim1 lo
    g1[3] = (int)((tensor_d1 >> 16) | (tile_d0 << 16));                  // dim1 hi | tile_d0
    g1[4] = (int)(tile_d1 & 0xffffu);                                    // tile_d1 | tile_d2=0
    g1[5] = (int)tensor_d0;                                              // dim0_stride lo32
    g1[6] = 0;
    g1[7] = 0;
    i32x4 z4 = {0, 0, 0, 0};
    i32x8 z8 = {0, 0, 0, 0, 0, 0, 0, 0};
    __builtin_amdgcn_tensor_load_to_lds(g0, g1, z4, z4, z8, 0);
}
#endif

// ==========================================================================
// GEMM: C[M,N] = act(A_bf16[M,K] @ Wt_bf16[N,K]^T + bias[N] (+ R[M,N]))
// 128x128x32 tiles, 256 threads (8 waves), wave owns 32x64 of C.
// Double-buffered LDS; TDM prefetches tile k+1 while tile k is computed.
// OMODE: 0 = fp32 C only, 1 = fp32 C + bf16 Cb, 2 = bf16 Cb only.
// ==========================================================================
template<bool RELU, bool RES, int OMODE>
__global__ __launch_bounds__(256)
void gemm_bias_kernel(const __bf16* __restrict__ A, const __bf16* __restrict__ Bt,
                      const float* __restrict__ bias, const float* __restrict__ R,
                      float* __restrict__ C, __bf16* __restrict__ Cb,
                      int M, int N, int K)
{
    constexpr int BM = 128, BN = 128, BK = 32, LDT = BK + 8;   // 80B rows
    __shared__ __bf16 As[2][BM * LDT];
    __shared__ __bf16 Bs[2][BN * LDT];   // [n][k]

    const int t    = threadIdx.x;
    const int lane = t & 31;
    const int w    = t >> 5;
    const int half = lane >> 4;
    const int col  = lane & 15;
    const int wm   = (w & 3) * 32;
    const int wn   = (w >> 2) * 64;
    const int m0   = blockIdx.y * BM;
    const int n0   = blockIdx.x * BN;

    v8f acc[2][4];
#pragma unroll
    for (int i = 0; i < 2; i++)
#pragma unroll
        for (int j = 0; j < 4; j++) acc[i][j] = zero8();

    const int nsteps = K / BK;

#if HAS_TDM
    const unsigned asOff[2] = { (unsigned)(size_t)&As[0][0], (unsigned)(size_t)&As[1][0] };
    const unsigned bsOff[2] = { (unsigned)(size_t)&Bs[0][0], (unsigned)(size_t)&Bs[1][0] };
    if (t < 32) {   // wave 0 drives the Tensor Data Mover; prologue: tile 0
        tdm_load_2d(asOff[0], A  + (size_t)m0 * K, K, M, BK, BM, 3u, 3u);
        tdm_load_2d(bsOff[0], Bt + (size_t)n0 * K, K, N, BK, BN, 3u, 3u);
    }
#endif

    for (int kt = 0; kt < nsteps; kt++) {
        const int cur = kt & 1;
#if HAS_TDM
        if (t < 32) {
            if (kt + 1 < nsteps) {   // prefetch next tile into the other buffer
                const int nxt = cur ^ 1;
                tdm_load_2d(asOff[nxt], A  + (size_t)m0 * K + (kt + 1) * BK, K, M, BK, BM, 3u, 3u);
                tdm_load_2d(bsOff[nxt], Bt + (size_t)n0 * K + (kt + 1) * BK, K, N, BK, BN, 3u, 3u);
                __builtin_amdgcn_s_wait_tensorcnt(2);   // tile kt has landed
            } else {
                __builtin_amdgcn_s_wait_tensorcnt(0);
            }
        }
#else
        {   // fallback: plain bf16 copies
            const int r16 = t >> 1, c16 = (t & 1) * 16;
            const __bf16* ga = A + (size_t)(m0 + r16) * K + kt * BK + c16;
            *(v8bf*)&As[cur][r16 * LDT + c16]     = *(const v8bf*)(ga);
            *(v8bf*)&As[cur][r16 * LDT + c16 + 8] = *(const v8bf*)(ga + 8);
            const __bf16* gb = Bt + (size_t)(n0 + r16) * K + kt * BK + c16;
            *(v8bf*)&Bs[cur][r16 * LDT + c16]     = *(const v8bf*)(gb);
            *(v8bf*)&Bs[cur][r16 * LDT + c16 + 8] = *(const v8bf*)(gb + 8);
        }
#endif
        __syncthreads();

        const int abase = half * 8;   // A lane K pattern {b..b+7, b+16..b+23}
        v16bf af[2], bfg[4];
#pragma unroll
        for (int mt = 0; mt < 2; mt++) {
            const __bf16* p = &As[cur][(wm + mt * 16 + col) * LDT];
            af[mt] = ld_frag(p + abase, p + abase + 16);
        }
#pragma unroll
        for (int nt = 0; nt < 4; nt++) {
            const __bf16* p = &Bs[cur][(wn + nt * 16 + col) * LDT + half * 8];
            bfg[nt] = ld_frag(p, p + 16);
        }
#pragma unroll
        for (int mt = 0; mt < 2; mt++)
#pragma unroll
            for (int nt = 0; nt < 4; nt++)
                acc[mt][nt] = WMMA_BF16(af[mt], bfg[nt], acc[mt][nt]);
        __syncthreads();
    }

    // ---- epilogue ----
#pragma unroll
    for (int mt = 0; mt < 2; mt++) {
#pragma unroll
        for (int nt = 0; nt < 4; nt++) {
            const int gn = n0 + wn + nt * 16 + col;
            const float bv = bias[gn];
#pragma unroll
            for (int r = 0; r < 8; r++) {
                const int gm = m0 + wm + mt * 16 + r + 8 * half;
                float v = acc[mt][nt][r] + bv;
                if (RES)  v += R[(size_t)gm * N + gn];
                if (RELU) v = fmaxf(v, 0.0f);
                if (OMODE != 2) C[(size_t)gm * N + gn]  = v;
                if (OMODE != 0) Cb[(size_t)gm * N + gn] = f2bf(v);
            }
        }
    }
}

// ==========================================================================
// Flash attention over one (batch, head). qkv is bf16 [4096, 3072]
// (q|k|v at col 0|1024|2048, head h at +h*64). out is bf16 [4096, 1024].
// 128 threads = 4 waves x 16 query rows; 32-key K/V tiles; online softmax.
// K tile double-buffered via TDM prefetch; scale applied to fp32 scores.
// ==========================================================================
__global__ __launch_bounds__(128)
void attention_kernel(const __bf16* __restrict__ qkv, __bf16* __restrict__ out)
{
    constexpr int LDK = THEAD_DIM + 8;  // 72 elems = 144B rows
    constexpr int LDV = 32 + 8;         // 40
    constexpr int LDP = 32 + 8;         // 40
    __shared__ __bf16 Ks[2][32 * LDK];  // [key][dh]
    __shared__ __bf16 Vt[THEAD_DIM * LDV]; // [dh][key]
    __shared__ __bf16 Ps[4 * 16 * LDP];

    const int t    = threadIdx.x;
    const int lane = t & 31;
    const int w    = t >> 5;
    const int half = lane >> 4;
    const int col  = lane & 15;
    const int b    = blockIdx.x >> 4;
    const int h    = blockIdx.x & 15;
    const int q0   = blockIdx.y * 64;

    const float scale = 0.03125f;   // DIM^-0.5
    const size_t rs3 = 3 * TDIM;
    const __bf16* qb = qkv + (size_t)(b * TSEQ) * rs3 + h * THEAD_DIM;
    const __bf16* kb = qb + TDIM;
    const __bf16* vb = qb + 2 * TDIM;

    // Q fragments straight from global bf16 (row = q0 + w*16 + col)
    const __bf16* qp = qb + (size_t)(q0 + w * 16 + col) * rs3;
    v16bf qf[2];
#pragma unroll
    for (int f = 0; f < 2; f++) {
        const __bf16* p0 = qp + f * 32 + half * 8;
        qf[f] = ld_frag(p0, p0 + 16);
    }

    float m_r[8], l_r[8];
    v8f o[4];
#pragma unroll
    for (int r = 0; r < 8; r++) { m_r[r] = -1e30f; l_r[r] = 0.0f; }
#pragma unroll
    for (int nt = 0; nt < 4; nt++) o[nt] = zero8();

    constexpr int NTILES = TSEQ / 32;

#if HAS_TDM
    const unsigned ksOff[2] = { (unsigned)(size_t)&Ks[0][0], (unsigned)(size_t)&Ks[1][0] };
    if (t < 32) {   // prologue: key tile 0
        tdm_load_2d(ksOff[0], kb, (unsigned)rs3, TROWS, THEAD_DIM, 32, 4u, 3u);
    }
#endif

    for (int kt = 0; kt < NTILES; kt++) {
        const int cur = kt & 1;
        const int key0 = kt * 32;
#if HAS_TDM
        if (t < 32) {
            if (kt + 1 < NTILES) {
                tdm_load_2d(ksOff[cur ^ 1], kb + (size_t)(key0 + 32) * rs3,
                            (unsigned)rs3, TROWS, THEAD_DIM, 32, 4u, 3u);
                __builtin_amdgcn_s_wait_tensorcnt(1);
            } else {
                __builtin_amdgcn_s_wait_tensorcnt(0);
            }
        }
#else
        {
            const int key = t >> 2, d0 = (t & 3) * 16;
            const __bf16* gk = kb + (size_t)(key0 + key) * rs3 + d0;
            *(v8bf*)&Ks[cur][key * LDK + d0]     = *(const v8bf*)(gk);
            *(v8bf*)&Ks[cur][key * LDK + d0 + 8] = *(const v8bf*)(gk + 8);
        }
#endif
        {   // V tile manual transpose: [dh][key]
            const int key = t >> 2, d0 = (t & 3) * 16;
            const __bf16* gv = vb + (size_t)(key0 + key) * rs3 + d0;
            v8bf v0 = *(const v8bf*)(gv);
            v8bf v1 = *(const v8bf*)(gv + 8);
#pragma unroll
            for (int i = 0; i < 8; i++) {
                Vt[(d0 + i) * LDV + key]     = v0[i];
                Vt[(d0 + 8 + i) * LDV + key] = v1[i];
            }
        }
        __syncthreads();

        // ---- S = Q @ K^T (16 q x 32 keys, contraction dh=64) ----
        v8f s[2]; s[0] = zero8(); s[1] = zero8();
#pragma unroll
        for (int sub = 0; sub < 2; sub++) {
#pragma unroll
            for (int f = 0; f < 2; f++) {
                const __bf16* p = &Ks[cur][(sub * 16 + col) * LDK + f * 32 + half * 8];
                v16bf kf = ld_frag(p, p + 16);
                s[sub] = WMMA_BF16(qf[f], kf, s[sub]);
            }
        }

        // ---- online softmax ----
        __bf16* Psw = &Ps[w * 16 * LDP];
#pragma unroll
        for (int r = 0; r < 8; r++) {
            float a0 = s[0][r] * scale, a1 = s[1][r] * scale;
            float mx = fmaxf(a0, a1);
            mx = fmaxf(mx, __shfl_xor(mx, 1, 32));
            mx = fmaxf(mx, __shfl_xor(mx, 2, 32));
            mx = fmaxf(mx, __shfl_xor(mx, 4, 32));
            mx = fmaxf(mx, __shfl_xor(mx, 8, 32));
            float mnew = fmaxf(m_r[r], mx);
            float corr = __expf(m_r[r] - mnew);
            float p0 = __expf(a0 - mnew);
            float p1 = __expf(a1 - mnew);
            float sum = p0 + p1;
            sum += __shfl_xor(sum, 1, 32);
            sum += __shfl_xor(sum, 2, 32);
            sum += __shfl_xor(sum, 4, 32);
            sum += __shfl_xor(sum, 8, 32);
            l_r[r] = l_r[r] * corr + sum;
            m_r[r] = mnew;
#pragma unroll
            for (int nt = 0; nt < 4; nt++) o[nt][r] *= corr;
            Psw[(r + 8 * half) * LDP + col]      = f2bf(p0);
            Psw[(r + 8 * half) * LDP + 16 + col] = f2bf(p1);
        }
        __syncthreads();

        // ---- O += P @ V ----
        const __bf16* pp = &Psw[col * LDP];
        v16bf pf = ld_frag(pp + half * 8, pp + half * 8 + 16);
#pragma unroll
        for (int nt = 0; nt < 4; nt++) {
            const __bf16* vp = &Vt[(nt * 16 + col) * LDV + half * 8];
            v16bf vf = ld_frag(vp, vp + 16);
            o[nt] = WMMA_BF16(pf, vf, o[nt]);
        }
        __syncthreads();
    }

    // ---- normalize + bf16 store ----
    const int orow0 = b * TSEQ + q0 + w * 16;
#pragma unroll
    for (int r = 0; r < 8; r++) {
        const float inv = 1.0f / l_r[r];
        const int gm = orow0 + r + 8 * half;
#pragma unroll
        for (int nt = 0; nt < 4; nt++)
            out[(size_t)gm * TDIM + h * THEAD_DIM + nt * 16 + col] =
                f2bf(o[nt][r] * inv);
    }
}

// ==========================================================================
// LayerNorm rows of 1024 (fp32 in, bf16 out for the next GEMM).
// ==========================================================================
__global__ __launch_bounds__(256)
void layernorm_kernel(const float* __restrict__ x, const float* __restrict__ g,
                      const float* __restrict__ beta, __bf16* __restrict__ y)
{
    const int row = blockIdx.x;
    const int t = threadIdx.x;
    const float* xr = x + (size_t)row * TDIM;
    float v[4]; float s = 0.0f, s2 = 0.0f;
#pragma unroll
    for (int i = 0; i < 4; i++) {
        v[i] = xr[t + i * 256];
        s += v[i]; s2 += v[i] * v[i];
    }
#pragma unroll
    for (int m = 1; m < 32; m <<= 1) {
        s  += __shfl_xor(s,  m, 32);
        s2 += __shfl_xor(s2, m, 32);
    }
    __shared__ float ss[8], ss2[8];
    if ((t & 31) == 0) { ss[t >> 5] = s; ss2[t >> 5] = s2; }
    __syncthreads();
    s = 0.0f; s2 = 0.0f;
#pragma unroll
    for (int i = 0; i < 8; i++) { s += ss[i]; s2 += ss2[i]; }
    const float mu  = s * (1.0f / TDIM);
    const float var = s2 * (1.0f / TDIM) - mu * mu;
    const float rsd = rsqrtf(var + 1e-5f);
#pragma unroll
    for (int i = 0; i < 4; i++) {
        const int c = t + i * 256;
        y[(size_t)row * TDIM + c] = f2bf((v[i] - mu) * rsd * g[c] + beta[c]);
    }
}

// ==========================================================================
// Weight prep: W[K,N] fp32 -> Wt[N,K] bf16 (transposed for TDM row staging).
// ==========================================================================
__global__ __launch_bounds__(256)
void transpose_cvt_kernel(const float* __restrict__ W, __bf16* __restrict__ Wt, int N, int K)
{
    const size_t idx = (size_t)blockIdx.x * 256 + threadIdx.x;
    const int k = (int)(idx / N), n = (int)(idx % N);
    Wt[(size_t)n * K + k] = f2bf(W[idx]);
}

__global__ __launch_bounds__(256)
void cvt_kernel(const float* __restrict__ x, __bf16* __restrict__ y)
{
    const size_t i = (size_t)blockIdx.x * 256 + threadIdx.x;
    y[i] = f2bf(x[i]);
}

// ==========================================================================
// Host orchestration (graph-capture safe).
// ==========================================================================
extern "C" void kernel_launch(void* const* d_in, const int* in_sizes, int n_in,
                              void* d_out, int out_size, void* d_ws, size_t ws_size,
                              hipStream_t stream)
{
    (void)in_sizes; (void)n_in; (void)out_size; (void)ws_size;
    const float* x_in = (const float*)d_in[0];
    const float* Wqkv = (const float*)d_in[1];
    const float* bqkv = (const float*)d_in[2];
    const float* Wo   = (const float*)d_in[3];
    const float* bo   = (const float*)d_in[4];
    const float* ln_g = (const float*)d_in[5];
    const float* ln_b = (const float*)d_in[6];
    const float* W1   = (const float*)d_in[7];
    const float* b1   = (const float*)d_in[8];
    const float* W2   = (const float*)d_in[9];
    const float* b2   = (const float*)d_in[10];

    // workspace: xA, xB fp32 residual stream; bf16 activations + weight stage
    float*  xA    = (float*)d_ws;
    float*  xB    = xA + (size_t)TROWS * TDIM;
    __bf16* xbf   = (__bf16*)(xB + (size_t)TROWS * TDIM);         // [4096,1024]
    __bf16* bigbf = xbf + (size_t)TROWS * TDIM;                   // qkv [4096,3072] / h1 [4096,2048]
    __bf16* tmpbf = bigbf + (size_t)TROWS * 3 * TDIM;             // attn / ln out [4096,1024]
    __bf16* wt    = tmpbf + (size_t)TROWS * TDIM;                 // weight stage [3072,1024] max

    (void)hipMemcpyAsync(xA, x_in, (size_t)TROWS * TDIM * sizeof(float),
                         hipMemcpyDeviceToDevice, stream);
    cvt_kernel<<<dim3(TROWS * TDIM / 256), dim3(256), 0, stream>>>(x_in, xbf);

    const dim3 blk(256);
    for (int l = 0; l < TDEPTH; l++) {
        const float* Wq  = Wqkv + (size_t)l * TDIM * 3 * TDIM;
        const float* bq  = bqkv + (size_t)l * 3 * TDIM;
        const float* Wol = Wo   + (size_t)l * TDIM * TDIM;
        const float* bol = bo   + (size_t)l * TDIM;
        const float* gl  = ln_g + (size_t)l * TDIM;
        const float* bl  = ln_b + (size_t)l * TDIM;
        const float* W1l = W1   + (size_t)l * TDIM * THID;
        const float* b1l = b1   + (size_t)l * THID;
        const float* W2l = W2   + (size_t)l * THID * TDIM;
        const float* b2l = b2   + (size_t)l * TDIM;

        // qkv = xbf @ Wqkv + bqkv   (bf16 out)
        transpose_cvt_kernel<<<dim3(TDIM * 3 * TDIM / 256), blk, 0, stream>>>(Wq, wt, 3 * TDIM, TDIM);
        gemm_bias_kernel<false, false, 2><<<dim3(3 * TDIM / 128, TROWS / 128), blk, 0, stream>>>(
            xbf, wt, bq, nullptr, nullptr, bigbf, TROWS, 3 * TDIM, TDIM);
        // attention (bf16 in/out)
        attention_kernel<<<dim3(TB * THEADS, TSEQ / 64), dim3(128), 0, stream>>>(bigbf, tmpbf);
        // xB = attn @ Wo + bo + xA   (fp32 out)
        transpose_cvt_kernel<<<dim3(TDIM * TDIM / 256), blk, 0, stream>>>(Wol, wt, TDIM, TDIM);
        gemm_bias_kernel<false, true, 0><<<dim3(TDIM / 128, TROWS / 128), blk, 0, stream>>>(
            tmpbf, wt, bol, xA, xB, nullptr, TROWS, TDIM, TDIM);
        // tmpbf = LN(xB)   (bf16 out)
        layernorm_kernel<<<dim3(TROWS), blk, 0, stream>>>(xB, gl, bl, tmpbf);
        // h1 = relu(ln @ W1 + b1)   (bf16 out)
        transpose_cvt_kernel<<<dim3(TDIM * THID / 256), blk, 0, stream>>>(W1l, wt, THID, TDIM);
        gemm_bias_kernel<true, false, 2><<<dim3(THID / 128, TROWS / 128), blk, 0, stream>>>(
            tmpbf, wt, b1l, nullptr, nullptr, bigbf, TROWS, THID, TDIM);
        // xA = h1 @ W2 + b2 + xB    (fp32 out + bf16 copy for next layer)
        transpose_cvt_kernel<<<dim3(THID * TDIM / 256), blk, 0, stream>>>(W2l, wt, TDIM, THID);
        gemm_bias_kernel<false, true, 1><<<dim3(TDIM / 128, TROWS / 128), blk, 0, stream>>>(
            bigbf, wt, b2l, xB, xA, xbf, TROWS, TDIM, THID);
    }

    (void)hipMemcpyAsync(d_out, xA, (size_t)TROWS * TDIM * sizeof(float),
                         hipMemcpyDeviceToDevice, stream);
}
